// EdgeConv2d_43413529428127
// MI455X (gfx1250) — matrix operand order repaired and checked by reading, not compiled, not run on hardware
//
#include <hip/hip_runtime.h>
#include <math.h>

// ---- problem constants (match reference) ----
#define NB    4        // batches
#define NC    96       // input channels
#define NOUT  192      // output channels
#define NN    10000    // nodes
#define NK    16       // neighbors
#define TWO_C 192      // 2*NC = GEMM K dimension

#define WROWS    96            // weight rows staged per half
#define WPITCH   200           // padded row pitch in halves (192 + 8) = 400 B
#define WPITCH4  25            // pitch in 16-byte units
#define WU4      24            // 16-byte units of real data per row

typedef __attribute__((ext_vector_type(16))) _Float16 v16h;
typedef __attribute__((ext_vector_type(8)))  float    v8f;

// plain 16-byte aligned POD for b128 loads (HIP's uint4 has ctors -> keep unions trivial)
struct __attribute__((aligned(16))) U4 { unsigned int x, y, z, w; };
union Frag { v16h h; U4 q[2]; };

// ---------------------------------------------------------------------------
// Preprocess 1: weight fp32 [192][192] -> f16 (row-major == A-matrix friendly)
// ---------------------------------------------------------------------------
__global__ void cvt_weight(const float* __restrict__ w, _Float16* __restrict__ wf, int n) {
    int i = blockIdx.x * blockDim.x + threadIdx.x;
    if (i < n) wf[i] = (_Float16)w[i];
}

// ---------------------------------------------------------------------------
// Preprocess 2: x fp32 [B][C][N] -> xT f16 [B][N][C]  (LDS tile transpose)
// one gathered node row becomes 96 contiguous f16 (192 B)
// ---------------------------------------------------------------------------
__global__ __launch_bounds__(256)
void xpose(const float* __restrict__ x, _Float16* __restrict__ xT) {
    __shared__ float tile[32][33];
    int b  = blockIdx.z;
    int c0 = blockIdx.y * 32;       // C = 96 -> 3 exact tiles
    int n0 = blockIdx.x * 32;       // N = 10000 -> 313 tiles (last partial)
    int tx = threadIdx.x, ty = threadIdx.y;   // 32 x 8
    #pragma unroll
    for (int i = 0; i < 4; ++i) {
        int c = c0 + ty + 8 * i;
        int n = n0 + tx;
        float v = 0.0f;
        if (n < NN) v = x[((size_t)b * NC + c) * NN + n];
        tile[ty + 8 * i][tx] = v;
    }
    __syncthreads();
    #pragma unroll
    for (int i = 0; i < 4; ++i) {
        int n = n0 + ty + 8 * i;
        int c = c0 + tx;
        if (n < NN) xT[((size_t)b * NN + n) * NC + c] = (_Float16)tile[tx][ty + 8 * i];
    }
}

// ---------------------------------------------------------------------------
// Main kernel: one workgroup = (batch, 16-node tile); 16 waves, wave w owns
// node w.  Wave GEMM: M=192 (12 tiles) x N=16 (the node's 16 neighbors) x
// K=192 (6 steps of v_wmma_f32_16x16x32_f16).  The f16 weight is staged in
// LDS in two 96-row halves (38.4 KB each) so all 16 waves share one
// global/L2 fetch; max over K folds into a 16-lane butterfly on the WMMA
// C layout.
// ---------------------------------------------------------------------------
__global__ __launch_bounds__(512)
void edgeconv_wmma(const _Float16* __restrict__ wf,     // [192][192] f16
                   const _Float16* __restrict__ xT,     // [B][N][96] f16
                   const long long* __restrict__ edge,  // [2][B][N][K] i64
                   const float* __restrict__ bias,      // [192]
                   float* __restrict__ out)             // [B][192][N]
{
    __shared__ _Float16 swf[WROWS * WPITCH];   // 38,400 B staged weight half
    __shared__ float    sbias[NOUT];

    const int t = threadIdx.x;
    if (t < NOUT) sbias[t] = bias[t];

    const int wave = t >> 5;
    const int lane = t & 31;
    const int hi   = (lane >= 16) ? 1 : 0;   // which 16-lane half
    const int kcol = lane & 15;              // neighbor index = WMMA column

    const int bt   = blockIdx.x;
    const int b    = bt / (NN / 16);
    const int tile = bt % (NN / 16);
    const int node = tile * 16 + wave;

    // ---- gather indices (int64) ----
    long long eoff = (((long long)b * NN + node) * NK + kcol);
    int idx_j = (int)edge[eoff];                                   // edge_index[0]
    int idx_i = (int)edge[(long long)NB * NN * NK + eoff];         // edge_index[1]

    const U4* xi_base = (const U4*)(xT + ((size_t)b * NN + idx_i) * NC);
    const U4* xj_base = (const U4*)(xT + ((size_t)b * NN + idx_j) * NC);

    // ---- build 6 B fragments in-register ----
    // B k order used here (contiguous, same as A): lane<16 holds K=32s..32s+15 (=x chunk 2s),
    // lane>=16 holds K=32s+16..32s+31 (=x chunk 2s+1). chunk = 16 halves = 2xU4.
    // s=0..2: feat = x_i ;  s=3..5: feat = x_j - x_i  (packed f16 sub)
    Frag bfr[6];
    #pragma unroll
    for (int s = 0; s < 3; ++s) {
        int c = 2 * s + hi;
        bfr[s].q[0] = xi_base[2 * c];
        bfr[s].q[1] = xi_base[2 * c + 1];
    }
    #pragma unroll
    for (int s = 0; s < 3; ++s) {
        int c = 2 * s + hi;
        Frag xj;
        xj.q[0] = xj_base[2 * c];
        xj.q[1] = xj_base[2 * c + 1];
        bfr[3 + s].h = xj.h - bfr[s].h;      // elementwise f16 vector sub
    }

    // ---- two staged weight halves: rows [0,96) then [96,192) ----
    const U4* wf4  = (const U4*)wf;
    U4*       swf4 = (U4*)swf;

    #pragma unroll 1
    for (int half = 0; half < 2; ++half) {
        __syncthreads();                      // previous reads done / bias visible
        // cooperative stage: 96 rows x 24 U4 -> padded pitch of 25 U4
        for (int i = t; i < WROWS * WU4; i += 512) {
            int row = i / WU4, col = i % WU4;
            swf4[row * WPITCH4 + col] = wf4[(size_t)(half * WROWS + row) * WU4 + col];
        }
        __syncthreads();

        #pragma unroll 2
        for (int m = 0; m < 6; ++m) {
            v8f acc = {};
            const char* wrow = (const char*)swf + (size_t)(m * 16 + kcol) * (WPITCH * 2);
            #pragma unroll
            for (int s = 0; s < 6; ++s) {
                // A k order used here (contiguous): lane<16: K 32s..32s+15;
                // lane>=16: K 32s+16..32s+31 (matches B).
                Frag a;
                int off1 = 64 * s + (hi ? 32 : 0);
                a.q[0] = *(const U4*)(wrow + off1);
                a.q[1] = *(const U4*)(wrow + off1 + 16);
                acc = __builtin_amdgcn_wmma_f32_16x16x32_f16(
                        false, a.h, false, bfr[s].h, (short)0, acc, false, false);
            }

            // ---- bias + erf-GELU + max over the 16 neighbor columns ----
            int chbase = half * WROWS + m * 16 + (hi ? 8 : 0);
            #pragma unroll
            for (int r = 0; r < 8; ++r) {
                float v = acc[r] + sbias[chbase + r];
                v = 0.5f * v * (1.0f + erff(v * 0.70710678118654752f));
                #pragma unroll
                for (int off = 8; off >= 1; off >>= 1)
                    v = fmaxf(v, __shfl_xor(v, off, 16));   // reduce across columns
                if (kcol == 0)
                    out[((size_t)b * NOUT + (chbase + r)) * NN + node] = v;
            }
        }
    }
}

// ---------------------------------------------------------------------------
extern "C" void kernel_launch(void* const* d_in, const int* in_sizes, int n_in,
                              void* d_out, int out_size, void* d_ws, size_t ws_size,
                              hipStream_t stream) {
    const float*     x    = (const float*)d_in[0];
    const long long* edge = (const long long*)d_in[1];
    const float*     w    = (const float*)d_in[2];
    const float*     bias = (const float*)d_in[3];
    float*           out  = (float*)d_out;

    // scratch layout: [0, 73728): weight f16 ; then xT f16 (7.68 MB)
    _Float16* wf = (_Float16*)d_ws;
    _Float16* xT = (_Float16*)((char*)d_ws + (size_t)NOUT * TWO_C * sizeof(_Float16));

    cvt_weight<<<(NOUT * TWO_C + 255) / 256, 256, 0, stream>>>(w, wf, NOUT * TWO_C);

    dim3 gx((NN + 31) / 32, NC / 32, NB);
    xpose<<<gx, dim3(32, 8), 0, stream>>>(x, xT);

    edgeconv_wmma<<<NB * (NN / 16), 512, 0, stream>>>(wf, xT, edge, bias, out);
}
